// OtaMatcher_51153060495642
// MI455X (gfx1250) — compile-verified, matrix operand order checked by reading, not run on hardware
//
#include <hip/hip_runtime.h>
#include <hip/hip_bf16.h>

// ---- problem constants (from reference) ----
#define BS    16
#define NQ    1000
#define NC    80
#define NG    200
#define NROWS (NG + 1)          // 201 rows incl. background
#define RPAD  208               // rows padded to 13 * 16
#define QPAD  1024              // cols padded to 32 * 32
#define CPAD  1e30f             // pad cost -> exp underflows to 0
#define EPS_F 0.1f
#define INV_EPS 10.0f
#define ITERS_ 100
#define ALPHA_ 0.25f
#define CC_ 2.0f
#define CB_ 5.0f
#define CG_ 2.0f
#define CBG_ 1.0f
// fg = K - 0.5*(T_ITERS - T_CUR) = 4 ; mu_fg = NQ - fg*NG = 200

typedef __attribute__((ext_vector_type(16))) _Float16 v16h;
typedef __attribute__((ext_vector_type(8)))  float    v8f;

__device__ __forceinline__ float softplus_f(float z) {
    return fmaxf(z, 0.0f) + log1pf(__expf(-fabsf(z)));   // log(1+exp(z)) stable
}

// ---------------------------------------------------------------------------
// Kernel A: per-(b,q) focal-loss precompute.
// ---------------------------------------------------------------------------
__global__ void ota_cls_pre_kernel(const float* __restrict__ logits,
                                   float* __restrict__ delta,
                                   float* __restrict__ negsum) {
    const int idx = blockIdx.x * blockDim.x + threadIdx.x;   // b*NQ + q
    if (idx >= BS * NQ) return;
    const float* L = logits + (size_t)idx * NC;
    float*       D = delta  + (size_t)idx * NC;
    float ns = 0.0f;
    #pragma unroll 4
    for (int c = 0; c < NC; ++c) {
        const float x   = L[c];
        const float p   = 1.0f / (1.0f + __expf(-x));
        const float pos = ALPHA_          * softplus_f(-x) * (1.0f - p) * (1.0f - p);
        const float neg = (1.0f - ALPHA_) * softplus_f( x) * p * p;
        D[c] = pos - neg;
        ns  += neg;
    }
    negsum[idx] = ns;
}

// ---------------------------------------------------------------------------
// Kernel B: build padded cost matrix C[b, 208, 1024].
//   gi == NG         -> background row
//   gi >= NROWS or q >= NQ -> CPAD
// ---------------------------------------------------------------------------
__global__ void ota_cost_kernel(const float* __restrict__ pred_boxes,
                                const int*   __restrict__ gt_classes,
                                const float* __restrict__ gt_boxes,
                                const float* __restrict__ img_sz,
                                const float* __restrict__ delta,
                                const float* __restrict__ negsum,
                                float* __restrict__ Cmat) {
    const int bg = blockIdx.x;            // 0 .. BS*RPAD-1
    const int b  = bg / RPAD;
    const int gi = bg % RPAD;
    float* Crow = Cmat + ((size_t)b * RPAD + gi) * QPAD;

    if (gi >= NROWS) {                    // padding rows
        for (int q = threadIdx.x; q < QPAD; q += blockDim.x) Crow[q] = CPAD;
        return;
    }

    const float i0 = img_sz[b * 4 + 0], i1 = img_sz[b * 4 + 1];
    const float i2 = img_sz[b * 4 + 2], i3 = img_sz[b * 4 + 3];
    const float r0 = 1.0f / i0, r1 = 1.0f / i1, r2i = 1.0f / i2, r3 = 1.0f / i3;

    float t0 = 0.f, t1 = 0.f, t2 = 0.f, t3 = 0.f, area_a = 0.f;
    float tb0 = 0.f, tb1 = 0.f, tb2 = 0.f, tb3 = 0.f;
    int cls = 0;
    if (gi < NG) {
        const float* tb = gt_boxes + ((size_t)b * NG + gi) * 4;
        t0 = tb[0]; t1 = tb[1]; t2 = tb[2]; t3 = tb[3];
        tb0 = t0 * r0; tb1 = t1 * r1; tb2 = t2 * r2i; tb3 = t3 * r3;
        area_a = (t2 - t0) * (t3 - t1);
        cls = gt_classes[b * NG + gi];
    }

    for (int q = threadIdx.x; q < QPAD; q += blockDim.x) {
        float cval;
        if (q >= NQ) {
            cval = CPAD;                                      // padding columns
        } else if (gi == NG) {
            cval = CBG_ * negsum[b * NQ + q];                 // background row
        } else {
            const float nsv = negsum[b * NQ + q];
            const float* ob = pred_boxes + ((size_t)b * NQ + q) * 4;
            const float o0 = ob[0], o1 = ob[1], o2 = ob[2], o3 = ob[3];
            const float l1 = fabsf(tb0 - o0 * r0)  + fabsf(tb1 - o1 * r1)
                           + fabsf(tb2 - o2 * r2i) + fabsf(tb3 - o3 * r3);
            const float area_b = (o2 - o0) * (o3 - o1);
            const float w  = fmaxf(fminf(t2, o2) - fmaxf(t0, o0), 0.0f);
            const float h  = fmaxf(fminf(t3, o3) - fmaxf(t1, o1), 0.0f);
            const float inter = w * h;
            const float uni   = area_a + area_b - inter;
            const float iou   = inter / uni;
            const float wc = fmaxf(fmaxf(t2, o2) - fminf(t0, o0), 0.0f);
            const float hc = fmaxf(fmaxf(t3, o3) - fminf(t1, o1), 0.0f);
            const float areac = wc * hc;
            const float giou  = iou - (areac - uni) / areac;
            const float lcls = nsv + delta[((size_t)b * NQ + q) * NC + cls];
            cval = CC_ * lcls + CB_ * l1 + CG_ * (-giou);
        }
        Crow[q] = cval;
    }
}

// ---------------------------------------------------------------------------
// Kernel C: fused log-domain Sinkhorn (100 iters) + pi + argmax.
// One 1024-thread workgroup (32 waves) per image; padded C stays L2-resident.
// u-update row sums use V_WMMA_F32_16X16X32_F16 with unguarded b128 feeds.
// ---------------------------------------------------------------------------
__global__ __launch_bounds__(1024) void ota_sinkhorn_kernel(
        const float* __restrict__ Cmat,
        float* __restrict__ piOut,
        float* __restrict__ indsOut) {
    const int b = blockIdx.x;
    const float*  C  = Cmat + (size_t)b * RPAD * QPAD;
    const float4* C4 = reinterpret_cast<const float4*>(C);

    __shared__ float u_s[RPAD];
    __shared__ __align__(16) float v_s[QPAD];
    __shared__ float rm_s[RPAD];
    const float4* V4 = reinterpret_cast<const float4*>(v_s);

    const int tid  = threadIdx.x;
    const int lane = tid & 31;
    const int wave = tid >> 5;

    if (tid < RPAD) u_s[tid] = 1.0f;     // initial u = ones
    v_s[tid] = 1.0f;                     // initial v = ones (pads too)
    __syncthreads();

    const float log_nu    = __logf(1.0f + 1e-8f);
    const float log_mu_fg = __logf(200.0f + 1e-8f);
    const float log_mu_bg = log_nu;

    v16h ones;
    #pragma unroll
    for (int e = 0; e < 16; ++e) ones[e] = (_Float16)1.0f;

    const int rowbase = wave * 16;       // waves 0..12 own 16-row tiles
    const int r = lane & 15;             // A-matrix row owned by this lane
    const int h = lane >> 4;             // K half-select (documented A layout)
    const int g_mine  = rowbase + r;     // 0..207 (pads handled by CPAD)
    const int rowoff4 = g_mine * (QPAD / 4);

    for (int it = 0; it < ITERS_; ++it) {
        // ---- v update: branchless two-pass logsumexp over rows, per column ----
        if (tid < NQ) {
            const int q = tid;
            const float vq = v_s[q];
            float m = -1e30f;
            for (int g = 0; g < NROWS; ++g)
                m = fmaxf(m, (u_s[g] - C[g * QPAD + q]) * INV_EPS);
            float s = 0.0f;
            for (int g = 0; g < NROWS; ++g)
                s += __expf((u_s[g] - C[g * QPAD + q]) * INV_EPS - m);
            const float ls = m + __logf(s) + vq * INV_EPS;
            v_s[q] = EPS_F * (log_nu - ls) + vq;
        }
        __syncthreads();

        // ---- u update: per 16-row tile, WMMA row-sum of exp ----
        if (rowbase < NROWS) {
            // pass 1: per-row max of t = (v[q]-C)/eps (pads give -1e31, inert)
            for (int rr = 0; rr < 16; ++rr) {
                const int g = rowbase + rr;
                const float4* Crow4 = C4 + g * (QPAD / 4);
                float m = -1e30f;
                #pragma unroll 2
                for (int c4 = lane; c4 < QPAD / 4; c4 += 32) {
                    const float4 cv = Crow4[c4];
                    const float4 vv = V4[c4];
                    m = fmaxf(m, (vv.x - cv.x) * INV_EPS);
                    m = fmaxf(m, (vv.y - cv.y) * INV_EPS);
                    m = fmaxf(m, (vv.z - cv.z) * INV_EPS);
                    m = fmaxf(m, (vv.w - cv.w) * INV_EPS);
                }
                #pragma unroll
                for (int off = 16; off > 0; off >>= 1)
                    m = fmaxf(m, __shfl_xor(m, off, 32));
                if (lane == 0) rm_s[g] = m;       // same-wave LDS, in-order
            }
            const float rmax = rm_s[g_mine];

            // pass 2: rowsum_q exp(t - rowmax) via WMMA (A = data, B = ones).
            // Lane's 16 A-elements are cols [q0+8h .. +7] and [q0+16+8h .. +7]
            // -> 4x b128 from C + 4x b128 from LDS per chunk, no guards.
            v8f acc = {};
            for (int cb = 0; cb < 32; ++cb) {
                const int qa = (cb * 32 + 8 * h) >> 2;    // float4 index, group A
                const int qb = qa + 4;                    // +16 cols, group B
                const float4 cA0 = C4[rowoff4 + qa],     cA1 = C4[rowoff4 + qa + 1];
                const float4 cB0 = C4[rowoff4 + qb],     cB1 = C4[rowoff4 + qb + 1];
                const float4 vA0 = V4[qa], vA1 = V4[qa + 1];
                const float4 vB0 = V4[qb], vB1 = V4[qb + 1];
                v16h a;
                a[0]  = (_Float16)__expf((vA0.x - cA0.x) * INV_EPS - rmax);
                a[1]  = (_Float16)__expf((vA0.y - cA0.y) * INV_EPS - rmax);
                a[2]  = (_Float16)__expf((vA0.z - cA0.z) * INV_EPS - rmax);
                a[3]  = (_Float16)__expf((vA0.w - cA0.w) * INV_EPS - rmax);
                a[4]  = (_Float16)__expf((vA1.x - cA1.x) * INV_EPS - rmax);
                a[5]  = (_Float16)__expf((vA1.y - cA1.y) * INV_EPS - rmax);
                a[6]  = (_Float16)__expf((vA1.z - cA1.z) * INV_EPS - rmax);
                a[7]  = (_Float16)__expf((vA1.w - cA1.w) * INV_EPS - rmax);
                a[8]  = (_Float16)__expf((vB0.x - cB0.x) * INV_EPS - rmax);
                a[9]  = (_Float16)__expf((vB0.y - cB0.y) * INV_EPS - rmax);
                a[10] = (_Float16)__expf((vB0.z - cB0.z) * INV_EPS - rmax);
                a[11] = (_Float16)__expf((vB0.w - cB0.w) * INV_EPS - rmax);
                a[12] = (_Float16)__expf((vB1.x - cB1.x) * INV_EPS - rmax);
                a[13] = (_Float16)__expf((vB1.y - cB1.y) * INV_EPS - rmax);
                a[14] = (_Float16)__expf((vB1.z - cB1.z) * INV_EPS - rmax);
                a[15] = (_Float16)__expf((vB1.w - cB1.w) * INV_EPS - rmax);
                acc = __builtin_amdgcn_wmma_f32_16x16x32_f16(
                        false, a, false, ones, (short)0, acc, false, false);
            }
            // D layout: lanes 0-15 vgpr j -> M=j ; lanes 16-31 vgpr j -> M=8+j
            if (lane == 0 || lane == 16) {
                #pragma unroll
                for (int r2 = 0; r2 < 8; ++r2) {
                    const int gg = rowbase + (lane >> 4) * 8 + r2;
                    if (gg < NROWS) {
                        const float s  = fmaxf(acc[r2], 1e-30f);
                        const float ls = rm_s[gg] + __logf(s) + u_s[gg] * INV_EPS;
                        const float lm = (gg == NG) ? log_mu_bg : log_mu_fg;
                        u_s[gg] = EPS_F * (lm - ls) + u_s[gg];
                    }
                }
            }
        }
        __syncthreads();
    }

    // ---- finalize: rm_s[g] = max_q (v[q]-C)/eps (pi row-max in log space) ----
    if (rowbase < NROWS) {
        for (int rr = 0; rr < 16; ++rr) {
            const int g = rowbase + rr;
            const float4* Crow4 = C4 + g * (QPAD / 4);
            float m = -1e30f;
            #pragma unroll 2
            for (int c4 = lane; c4 < QPAD / 4; c4 += 32) {
                const float4 cv = Crow4[c4];
                const float4 vv = V4[c4];
                m = fmaxf(m, (vv.x - cv.x) * INV_EPS);
                m = fmaxf(m, (vv.y - cv.y) * INV_EPS);
                m = fmaxf(m, (vv.z - cv.z) * INV_EPS);
                m = fmaxf(m, (vv.w - cv.w) * INV_EPS);
            }
            #pragma unroll
            for (int off = 16; off > 0; off >>= 1)
                m = fmaxf(m, __shfl_xor(m, off, 32));
            if (lane == 0) rm_s[g] = m;
        }
    }
    __syncthreads();

    // ---- pi = exp(M - rowmaxM) and per-column argmax ----
    if (tid < NQ) {
        const int q = tid;
        float best = -1.0f;
        int   bidx = 0;
        float* po = piOut + (size_t)b * NROWS * NQ;
        const float vq = v_s[q];
        for (int g = 0; g < NROWS; ++g) {
            const float val = __expf((vq - C[g * QPAD + q]) * INV_EPS - rm_s[g]);
            po[g * NQ + q] = val;
            if (val > best) { best = val; bidx = g; }   // first max wins
        }
        indsOut[b * NQ + q] = (float)bidx;
    }
}

// ---------------------------------------------------------------------------
// Launch
// ---------------------------------------------------------------------------
extern "C" void kernel_launch(void* const* d_in, const int* in_sizes, int n_in,
                              void* d_out, int out_size, void* d_ws, size_t ws_size,
                              hipStream_t stream) {
    (void)in_sizes; (void)n_in; (void)out_size; (void)ws_size;

    const float* pred_logits = (const float*)d_in[0];   // [16,1000,80]
    const float* pred_boxes  = (const float*)d_in[1];   // [16,1000,4]
    const int*   gt_classes  = (const int*)  d_in[2];   // [16,200]
    const float* gt_boxes    = (const float*)d_in[3];   // [16,200,4]
    const float* img_sz      = (const float*)d_in[4];   // [16,4]
    // d_in[5] image_size_xyxy_tgt: rows identical to img_sz -> unused

    float* out  = (float*)d_out;
    float* pi   = out;                                  // [16,201,1000]
    float* inds = out + (size_t)BS * NROWS * NQ;        // [16,1000] (as float)

    // workspace carve-out (~18.8 MB total)
    char*  ws     = (char*)d_ws;
    float* delta  = (float*)ws;                              // 16*1000*80*4  = 5,120,000 B
    float* negsum = (float*)(ws + 5120000);                  // 16*1000*4     =    64,000 B
    float* Cmat   = (float*)(ws + 5184000);                  // 16*208*1024*4 = 13,631,488 B

    ota_cls_pre_kernel<<<(BS * NQ + 255) / 256, 256, 0, stream>>>(
        pred_logits, delta, negsum);

    ota_cost_kernel<<<BS * RPAD, 256, 0, stream>>>(
        pred_boxes, gt_classes, gt_boxes, img_sz, delta, negsum, Cmat);

    ota_sinkhorn_kernel<<<BS, 1024, 0, stream>>>(Cmat, pi, inds);
}
